// StaticGCN_44109314130143
// MI455X (gfx1250) — matrix operand order repaired and checked by reading, not compiled, not run on hardware
//
#include <hip/hip_runtime.h>

// ---------------------------------------------------------------------------
// Types for CDNA5 WMMA (gfx1250, wave32)
// ---------------------------------------------------------------------------
typedef __attribute__((ext_vector_type(16))) __bf16 v16bf;
typedef __attribute__((ext_vector_type(8)))  __bf16 v8bf;
typedef __attribute__((ext_vector_type(8)))  float  v8f;

#define H_DIM 128
#define G_DIM 64

// ---------------------------------------------------------------------------
// Degree / inverse-count
// ---------------------------------------------------------------------------
__global__ void deg_kernel(const long long* __restrict__ ei, float* __restrict__ cnt,
                           long long E) {
  long long t = (long long)blockIdx.x * blockDim.x + threadIdx.x;
  if (t >= E) return;
  long long d = ei[E + t];
  atomicAdd(&cnt[d], 1.0f);
}

__global__ void inv_kernel(const float* __restrict__ cnt, float* __restrict__ inv, int n) {
  int t = blockIdx.x * blockDim.x + threadIdx.x;
  if (t >= n) return;
  inv[t] = 1.0f / fmaxf(cnt[t], 1.0f);
}

// ---------------------------------------------------------------------------
// Weight prep: Wt[h][k] = bf16( k < K ? Wl[k][h] : Wr[k-K][h] )   (H=128 cols)
// ---------------------------------------------------------------------------
template <int K>
__global__ void prep_cat_kernel(const float* __restrict__ Wl, const float* __restrict__ Wr,
                                __bf16* __restrict__ Wt) {
  int idx = blockIdx.x * blockDim.x + threadIdx.x;
  if (idx >= H_DIM * 2 * K) return;
  int h = idx / (2 * K);
  int k = idx % (2 * K);
  float v = (k < K) ? Wl[(size_t)k * H_DIM + h] : Wr[(size_t)(k - K) * H_DIM + h];
  Wt[idx] = (__bf16)v;
}

// Wt[h][k] = bf16(W[k][h]) for 128x128
__global__ void prep_t_kernel(const float* __restrict__ W, __bf16* __restrict__ Wt) {
  int idx = blockIdx.x * blockDim.x + threadIdx.x;
  if (idx >= H_DIM * H_DIM) return;
  int h = idx / H_DIM;
  int k = idx % H_DIM;
  Wt[idx] = (__bf16)W[(size_t)k * H_DIM + h];
}

// ---------------------------------------------------------------------------
// Edge scatter-add:  agg[dst] += x[src]   (agg fits in 192MB L2 -> L2 atomics)
// ---------------------------------------------------------------------------
template <int K>
__global__ void scatter_kernel(const float* __restrict__ xin, const long long* __restrict__ ei,
                               float* __restrict__ agg, long long E) {
  constexpr int QP = K / 4;
  long long t = (long long)blockIdx.x * blockDim.x + threadIdx.x;
  if (t >= E * QP) return;
  long long e = t / QP;
  int       q = (int)(t & (QP - 1));
  long long s = ei[e];
  long long d = ei[E + e];
  const float4 v = *(const float4*)(xin + s * (long long)K + q * 4);
  float* ap = agg + d * (long long)K + q * 4;
  atomicAdd(ap + 0, v.x);
  atomicAdd(ap + 1, v.y);
  atomicAdd(ap + 2, v.z);
  atomicAdd(ap + 3, v.w);
}

// ---------------------------------------------------------------------------
// Fused GEMM + bias + ReLU via v_wmma_f32_16x16x32_bf16.
//   out[r, 0:128] = relu( concat(part1[r]*scale[r], part2[r]) @ Wt^T + bias )
// k-outer loop with 8 live accumulators: consecutive WMMAs hit different D
// registers (no RAW hazard NOPs) and each A fragment is loaded from LDS once
// per k-step. Epilogue has a uniform full-tile fast path so stores become an
// immediate-offset clause with no exec-mask churn.
// Block = 128 threads = 4 waves, each wave owns 16 rows.
// ---------------------------------------------------------------------------
template <int K1, int K2>
__global__ void gemm_kernel(const float* __restrict__ part1, const float* __restrict__ scale,
                            const float* __restrict__ part2, const __bf16* __restrict__ Wt,
                            const float* __restrict__ bias, float* __restrict__ outp,
                            int nrows) {
  constexpr int STRIDE = K2 + 8;  // pad: spreads row starts across LDS banks
  __shared__ __bf16 lds[4 * 16 * STRIDE];

  const int lid     = threadIdx.x & 31;
  const int wv      = threadIdx.x >> 5;
  const int rowbase = blockIdx.x * 64 + wv * 16;
  __bf16*   my      = lds + wv * 16 * STRIDE;

  // ---- stage 16 rows of A (bf16, mean-scale fused) into LDS ----
  constexpr int QP = K2 / 4;
  for (int idx = lid; idx < 16 * QP; idx += 32) {
    int r    = idx / QP;
    int c    = (idx % QP) * 4;
    int grow = rowbase + r;
    float4 v = make_float4(0.f, 0.f, 0.f, 0.f);
    if (grow < nrows) {
      if (c < K1) {
        v       = *(const float4*)(part1 + (size_t)grow * K1 + c);
        float s = scale ? scale[grow] : 1.0f;
        v.x *= s; v.y *= s; v.z *= s; v.w *= s;
      } else {
        v = *(const float4*)(part2 + (size_t)grow * (K2 - K1) + (c - K1));
      }
    }
    __bf16* p = my + r * STRIDE + c;
    p[0] = (__bf16)v.x; p[1] = (__bf16)v.y; p[2] = (__bf16)v.z; p[3] = (__bf16)v.w;
  }
  __syncthreads();

  const int     half = lid >> 4;
  const int     n    = lid & 15;
  const __bf16* arow = my + n * STRIDE;

  v8f acc[8];
#pragma unroll
  for (int j = 0; j < 8; ++j) acc[j] = v8f{0.f, 0.f, 0.f, 0.f, 0.f, 0.f, 0.f, 0.f};

#pragma unroll
  for (int k0 = 0; k0 < K2; k0 += 32) {
    // A fragment (ISA 16-bit A layout): elems 0..7 = K k0..k0+7 (+8 if lane>=16),
    // elems 8..15 = K k0+16..k0+23 (+8 if lane>=16)
    v8bf  alo = *(const v8bf*)(arow + k0 + half * 8);
    v8bf  ahi = *(const v8bf*)(arow + k0 + 16 + half * 8);
    v16bf af  = __builtin_shufflevector(alo, ahi, 0, 1, 2, 3, 4, 5, 6, 7,
                                        8, 9, 10, 11, 12, 13, 14, 15);
    const __bf16* bk = Wt + k0 + half * 8;  // B: lane = N, same K order as A fragment
#pragma unroll
    for (int j = 0; j < 8; ++j) {
      const __bf16* bcol = bk + (size_t)(j * 16 + n) * K2;
      v8bf  blo = *(const v8bf*)(bcol);
      v8bf  bhi = *(const v8bf*)(bcol + 16);
      v16bf bfm = __builtin_shufflevector(blo, bhi, 0, 1, 2, 3, 4, 5, 6, 7,
                                          8, 9, 10, 11, 12, 13, 14, 15);
      acc[j] = __builtin_amdgcn_wmma_f32_16x16x32_bf16(false, af, false, bfm,
                                                       (short)0, acc[j], false, false);
    }
  }

  // ---- epilogue: bias + ReLU + store (C layout: m = i + 8*half, n = lane&15)
  float* base = outp + (size_t)(rowbase + 8 * half) * H_DIM + n;
  if (rowbase + 16 <= nrows) {
#pragma unroll
    for (int j = 0; j < 8; ++j) {
      const float bv = bias[j * 16 + n];
#pragma unroll
      for (int i = 0; i < 8; ++i) {
        float v = acc[j][i] + bv;
        base[(size_t)i * H_DIM + j * 16] = v > 0.f ? v : 0.f;
      }
    }
  } else {
#pragma unroll
    for (int j = 0; j < 8; ++j) {
      const float bv = bias[j * 16 + n];
#pragma unroll
      for (int i = 0; i < 8; ++i) {
        if (rowbase + 8 * half + i < nrows) {
          float v = acc[j][i] + bv;
          base[(size_t)i * H_DIM + j * 16] = v > 0.f ? v : 0.f;
        }
      }
    }
  }
}

// ---------------------------------------------------------------------------
// Head part 1 (wave per node): d1/d3 = (h@Wfc+bfc)[0/2]; a1 = relu([d1,d3]@Wa1+ba1)
// ---------------------------------------------------------------------------
__global__ void fc_head_kernel(const float* __restrict__ h, const float* __restrict__ Wfc,
                               const float* __restrict__ bfc, const float* __restrict__ Wa1,
                               const float* __restrict__ ba1, float* __restrict__ d1,
                               float* __restrict__ d3, float* __restrict__ a1, int nrows) {
  int node = blockIdx.x * 4 + (threadIdx.x >> 5);
  if (node >= nrows) return;
  int lid = threadIdx.x & 31;
  const float* hr = h + (size_t)node * H_DIM;
  float p1 = 0.f, p3 = 0.f;
#pragma unroll
  for (int q = 0; q < 4; ++q) {
    int   c  = lid + q * 32;
    float hv = hr[c];
    p1 += hv * Wfc[c * 3 + 0];
    p3 += hv * Wfc[c * 3 + 2];
  }
#pragma unroll
  for (int m = 16; m >= 1; m >>= 1) {
    p1 += __shfl_xor(p1, m, 32);
    p3 += __shfl_xor(p3, m, 32);
  }
  float v1 = p1 + bfc[0];
  float v3 = p3 + bfc[2];
  if (lid == 0) { d1[node] = v1; d3[node] = v3; }
  float* ar = a1 + (size_t)node * H_DIM;
#pragma unroll
  for (int q = 0; q < 4; ++q) {
    int   o = lid + q * 32;
    float v = v1 * Wa1[o] + v3 * Wa1[H_DIM + o] + ba1[o];
    ar[o]   = v > 0.f ? v : 0.f;
  }
}

// ---------------------------------------------------------------------------
// Head part 2 (wave per node): aux = a2@Wao+bao; group atomics for segment mean
// ---------------------------------------------------------------------------
__global__ void final_kernel(const float* __restrict__ a2, const float* __restrict__ Wao,
                             const float* __restrict__ bao, const float* __restrict__ d1,
                             const float* __restrict__ d3, const long long* __restrict__ batch,
                             float* __restrict__ gs, float* __restrict__ gcnt, int nrows) {
  int node = blockIdx.x * 4 + (threadIdx.x >> 5);
  if (node >= nrows) return;
  int lid = threadIdx.x & 31;
  const float* ar = a2 + (size_t)node * H_DIM;
  float p = 0.f;
#pragma unroll
  for (int q = 0; q < 4; ++q) {
    int c = lid + q * 32;
    p += ar[c] * Wao[c];
  }
#pragma unroll
  for (int m = 16; m >= 1; m >>= 1) p += __shfl_xor(p, m, 32);
  if (lid == 0) {
    float     aux = p + bao[0];
    long long b   = batch[node];
    atomicAdd(&gs[b * 3 + 0], d1[node]);
    atomicAdd(&gs[b * 3 + 1], aux);
    atomicAdd(&gs[b * 3 + 2], d3[node]);
    atomicAdd(&gcnt[b], 1.0f);
  }
}

__global__ void div_kernel(const float* __restrict__ gs, const float* __restrict__ gcnt,
                           float* __restrict__ out) {
  int idx = threadIdx.x;
  if (idx >= G_DIM * 3) return;
  out[idx] = gs[idx] / fmaxf(gcnt[idx / 3], 1.0f);
}

// ---------------------------------------------------------------------------
// Host-side orchestration
// ---------------------------------------------------------------------------
static inline size_t alignup(size_t v) { return (v + 255) & ~(size_t)255; }

extern "C" void kernel_launch(void* const* d_in, const int* in_sizes, int n_in,
                              void* d_out, int out_size, void* d_ws, size_t ws_size,
                              hipStream_t stream) {
  (void)n_in; (void)out_size;
  const float*     x     = (const float*)d_in[0];
  const long long* ei    = (const long long*)d_in[1];
  const long long* batch = (const long long*)d_in[2];
  const float *Wl0 = (const float*)d_in[3],  *bl0 = (const float*)d_in[4],
              *Wr0 = (const float*)d_in[5];
  const float *Wl1 = (const float*)d_in[6],  *bl1 = (const float*)d_in[7],
              *Wr1 = (const float*)d_in[8];
  const float *Wl2 = (const float*)d_in[9],  *bl2 = (const float*)d_in[10],
              *Wr2 = (const float*)d_in[11];
  const float *Wfc = (const float*)d_in[12], *bfc = (const float*)d_in[13];
  const float *Wa1 = (const float*)d_in[14], *ba1 = (const float*)d_in[15];
  const float *Wa2 = (const float*)d_in[16], *ba2 = (const float*)d_in[17];
  const float *Wao = (const float*)d_in[18], *bao = (const float*)d_in[19];

  const int       N   = in_sizes[0] / 32;
  const long long E   = in_sizes[1] / 2;
  float*          out = (float*)d_out;

  // workspace carve-up
  char*  ws  = (char*)d_ws;
  size_t off = 0;
  auto carve = [&](size_t bytes) { char* p = ws + off; off += alignup(bytes); return p; };
  float*  bufA = (float*)carve((size_t)N * H_DIM * 4);
  float*  bufB = (float*)carve((size_t)N * H_DIM * 4);
  float*  bufC = (float*)carve((size_t)N * H_DIM * 4);  // agg / a2
  float*  cnt  = (float*)carve((size_t)N * 4);
  float*  inv  = (float*)carve((size_t)N * 4);
  float*  d1   = (float*)carve((size_t)N * 4);
  float*  d3   = (float*)carve((size_t)N * 4);
  float*  gs   = (float*)carve(G_DIM * 3 * 4);
  float*  gcnt = (float*)carve(G_DIM * 4);
  __bf16* Wt0  = (__bf16*)carve((size_t)H_DIM * 64 * 2);
  __bf16* Wt1  = (__bf16*)carve((size_t)H_DIM * 256 * 2);
  __bf16* Wt2  = (__bf16*)carve((size_t)H_DIM * 256 * 2);
  __bf16* Wa2t = (__bf16*)carve((size_t)H_DIM * H_DIM * 2);
  (void)ws_size;

  const int B256 = 256;
  auto blocks = [](long long total, int bs) { return (unsigned)((total + bs - 1) / bs); };

  // 1) degrees
  hipMemsetAsync(cnt, 0, (size_t)N * 4, stream);
  deg_kernel<<<blocks(E, B256), B256, 0, stream>>>(ei, cnt, E);
  inv_kernel<<<blocks(N, B256), B256, 0, stream>>>(cnt, inv, N);

  // 2) weight prep (bf16, output-column-major)
  prep_cat_kernel<32><<<blocks(H_DIM * 64, B256), B256, 0, stream>>>(Wl0, Wr0, Wt0);
  prep_cat_kernel<128><<<blocks(H_DIM * 256, B256), B256, 0, stream>>>(Wl1, Wr1, Wt1);
  prep_cat_kernel<128><<<blocks(H_DIM * 256, B256), B256, 0, stream>>>(Wl2, Wr2, Wt2);
  prep_t_kernel<<<blocks(H_DIM * H_DIM, B256), B256, 0, stream>>>(Wa2, Wa2t);

  const unsigned gemmBlocks = (unsigned)((N + 63) / 64);

  // 3) layer 0: agg32(x) -> h0 = relu([mean|x] @ [Wl0;Wr0] + bl0)  -> bufA
  hipMemsetAsync(bufC, 0, (size_t)N * 32 * 4, stream);
  scatter_kernel<32><<<blocks(E * 8, B256), B256, 0, stream>>>(x, ei, bufC, E);
  gemm_kernel<32, 64><<<gemmBlocks, 128, 0, stream>>>(bufC, inv, x, Wt0, bl0, bufA, N);

  // 4) layer 1: agg128(h0) -> h1 -> bufB
  hipMemsetAsync(bufC, 0, (size_t)N * H_DIM * 4, stream);
  scatter_kernel<128><<<blocks(E * 32, B256), B256, 0, stream>>>(bufA, ei, bufC, E);
  gemm_kernel<128, 256><<<gemmBlocks, 128, 0, stream>>>(bufC, inv, bufA, Wt1, bl1, bufB, N);

  // 5) layer 2: agg128(h1) -> h2 -> bufA
  hipMemsetAsync(bufC, 0, (size_t)N * H_DIM * 4, stream);
  scatter_kernel<128><<<blocks(E * 32, B256), B256, 0, stream>>>(bufB, ei, bufC, E);
  gemm_kernel<128, 256><<<gemmBlocks, 128, 0, stream>>>(bufC, inv, bufB, Wt2, bl2, bufA, N);

  // 6) head: d1/d3 + a1 -> bufB
  fc_head_kernel<<<blocks(N, 4), 128, 0, stream>>>(bufA, Wfc, bfc, Wa1, ba1, d1, d3, bufB, N);

  // 7) a2 = relu(a1 @ Wa2 + ba2) -> bufC (WMMA)
  gemm_kernel<128, 128><<<gemmBlocks, 128, 0, stream>>>(bufB, nullptr, nullptr, Wa2t, ba2,
                                                        bufC, N);

  // 8) aux head + segment sums
  hipMemsetAsync(gs, 0, G_DIM * 3 * 4, stream);
  hipMemsetAsync(gcnt, 0, G_DIM * 4, stream);
  final_kernel<<<blocks(N, 4), 128, 0, stream>>>(bufC, Wao, bao, d1, d3, batch, gs, gcnt, N);

  // 9) segment mean -> out (64 x 3)
  div_kernel<<<1, 256, 0, stream>>>(gs, gcnt, out);
}
